// SimpleGNN_42210938585862
// MI455X (gfx1250) — matrix operand order repaired and checked by reading, not compiled, not run on hardware
//
#include <hip/hip_runtime.h>

typedef __attribute__((ext_vector_type(16))) _Float16 v16h;
typedef __attribute__((ext_vector_type(8)))  _Float16 v8h;
typedef __attribute__((ext_vector_type(8)))  float    v8f;

#define N_PART 1024
#define H_DIM  128
#define JCHUNK 128
#define ST     136   // padded LDS row stride in halves (16B pad -> conflict-free b128 frags)

__device__ __forceinline__ float fast_tanh(float x) {
#if __has_builtin(__builtin_amdgcn_tanhf)
  return __builtin_amdgcn_tanhf(x);      // v_tanh_f32 (gfx1250 trans op)
#elif __has_builtin(__builtin_amdgcn_tanh_f32)
  return __builtin_amdgcn_tanh_f32(x);
#else
  return tanhf(x);
#endif
}

__device__ __forceinline__ float gelu_t(float x) {
  // jax.nn.gelu approximate=True: 0.5x(1+tanh(sqrt(2/pi)(x+0.044715x^3)))
  const float k0 = 0.7978845608028654f;
  const float k1 = 0.044715f;
  float u = k0 * (x + k1 * x * x * x);
  return 0.5f * x * (1.0f + fast_tanh(u));
}

__global__ __launch_bounds__(256) void
gnn_pair_kernel(const float* __restrict__ xs,  const float* __restrict__ gs,
                const float* __restrict__ eW1, const float* __restrict__ eb1,
                const float* __restrict__ eW2, const float* __restrict__ eb2,
                const float* __restrict__ eW3, const float* __restrict__ eb3,
                const float* __restrict__ dW1, const float* __restrict__ db1,
                const float* __restrict__ dW2, const float* __restrict__ db2,
                const float* __restrict__ dW3, const float* __restrict__ db3,
                float* __restrict__ out)
{
  __shared__ __align__(16) _Float16 s_w2t [H_DIM * ST];   // W2 transposed [n][k], f16
  __shared__ __align__(16) _Float16 s_h1  [JCHUNK * ST];  // GELU(h1) tile [j][k], f16
  __shared__ __align__(16) _Float16 s_arow[JCHUNK * 8];   // per-j A row: dx,dy,gjx,gjy,1,0,0,0
  __shared__ __align__(16) _Float16 s_abf [H_DIM * 8];    // per-n B col: w0,w1,w4,w5,ai,0,0,0
  __shared__ __align__(16) _Float16 s_zero[8];            // zero slot for masked frag lanes
  __shared__ float s_b2  [H_DIM];
  __shared__ float s_pool[H_DIM];                         // sum_j gelu(h1@W2+b2)
  __shared__ float s_tmp [H_DIM];

  const int tid   = threadIdx.x;
  const int i     = blockIdx.x;
  const int lane  = tid & 31;
  const int wave  = tid >> 5;
  const int lmod  = lane & 15;
  const int lhalf = lane >> 4;

  const float xi0 = xs[2*i],   xi1 = xs[2*i+1];
  const float gi0 = gs[2*i],   gi1 = gs[2*i+1];

  // ---- stage weights into LDS ----
  for (int e = tid; e < H_DIM * H_DIM; e += 256) {
    int k = e >> 7, n = e & (H_DIM - 1);
    s_w2t[n * ST + k] = (_Float16)eW2[e];          // transpose: B[k][n] stored [n][k]
  }
  if (tid < H_DIM) {
    v8h w;
    w[0] = (_Float16)eW1[0*H_DIM + tid];
    w[1] = (_Float16)eW1[1*H_DIM + tid];
    w[2] = (_Float16)eW1[4*H_DIM + tid];
    w[3] = (_Float16)eW1[5*H_DIM + tid];
    w[4] = (_Float16)(gi0 * eW1[2*H_DIM + tid] + gi1 * eW1[3*H_DIM + tid] + eb1[tid]);
    w[5] = (_Float16)0.0f; w[6] = (_Float16)0.0f; w[7] = (_Float16)0.0f;
    *(v8h*)&s_abf[tid * 8] = w;
    s_b2[tid]   = eb2[tid];
    s_pool[tid] = 0.0f;
  }
  if (tid < 8) s_zero[tid] = (_Float16)0.0f;

  const v8h z8 = {};

  // ---- loop over neighbor chunks ----
  for (int jb = 0; jb < N_PART; jb += JCHUNK) {
    __syncthreads();   // previous chunk done with s_h1 / s_arow (also covers staging, 1st iter)

    if (tid < JCHUNK) {
      int j = jb + tid;
      float dx = xi0 - xs[2*j];
      dx -= 10.0f * rintf(dx * 0.1f);              // minimum-image wrap, RNE == jnp.round
      float dy = xi1 - xs[2*j+1];
      dy -= 10.0f * rintf(dy * 0.1f);
      v8h a;
      a[0] = (_Float16)dx;        a[1] = (_Float16)dy;
      a[2] = (_Float16)gs[2*j];   a[3] = (_Float16)gs[2*j+1];
      a[4] = (_Float16)1.0f;
      a[5] = (_Float16)0.0f; a[6] = (_Float16)0.0f; a[7] = (_Float16)0.0f;
      *(v8h*)&s_arow[tid * 8] = a;
    }
    __syncthreads();

    // ---- layer 1 via WMMA (K=5 padded to one K=32 step): h1pre = Arow @ Wpack ----
    {
      const _Float16* ap = lhalf ? s_zero : &s_arow[((wave << 4) + lmod) * 8];
      v8h a0 = *(const v8h*)ap;
      v16h av = __builtin_shufflevector(a0, z8, 0,1,2,3,4,5,6,7,8,9,10,11,12,13,14,15);
      v8f h1a[8];
      #pragma unroll
      for (int t = 0; t < 8; ++t) {
        const _Float16* bp = lhalf ? s_zero : &s_abf[((t << 4) + lmod) * 8];
        v8h b0 = *(const v8h*)bp;
        v16h bv = __builtin_shufflevector(b0, z8, 0,1,2,3,4,5,6,7,8,9,10,11,12,13,14,15);
        v8f cz = {};
        h1a[t] = __builtin_amdgcn_wmma_f32_16x16x32_f16(false, av, false, bv,
                                                        (short)0, cz, false, false);
      }
      // GELU + cvt + scatter into s_h1 (C layout: VGPR r -> m = r + 8*lhalf, n = 16t + lmod)
      #pragma unroll
      for (int t = 0; t < 8; ++t) {
        const int n = (t << 4) + lmod;
        #pragma unroll
        for (int r = 0; r < 8; ++r) {
          const int m = (wave << 4) + (lhalf << 3) + r;
          s_h1[m * ST + n] = (_Float16)gelu_t(h1a[t][r]);
        }
      }
    }
    __syncthreads();

    // ---- layer 2 GEMM: each wave computes rows [16w, 16w+16) x all 128 n via WMMA ----
    v8f acc[8] = {};
    const int mrow = (wave << 4) + lmod;
    #pragma unroll
    for (int ks = 0; ks < 4; ++ks) {
      const int k0 = ks << 5;
      // A frag (16x32 f16): lane holds row m=lane%16; k = {c..c+7, c+16..c+23}, c=(lane/16)*8
      const _Float16* ap = &s_h1[mrow * ST + k0 + (lhalf << 3)];
      v8h a0 = *(const v8h*)ap;
      v8h a1 = *(const v8h*)(ap + 16);
      v16h av = __builtin_shufflevector(a0, a1, 0,1,2,3,4,5,6,7,8,9,10,11,12,13,14,15);
      #pragma unroll
      for (int t = 0; t < 8; ++t) {
        // B frag (32x16 f16): lane holds col n=lane%16; k = {c..c+7, c+16..c+23}, c=(lane/16)*8
        const _Float16* bp = &s_w2t[((t << 4) + lmod) * ST + k0 + (lhalf << 3)];
        v8h b0 = *(const v8h*)bp;
        v8h b1 = *(const v8h*)(bp + 16);
        v16h bv = __builtin_shufflevector(b0, b1, 0,1,2,3,4,5,6,7,8,9,10,11,12,13,14,15);
        acc[t] = __builtin_amdgcn_wmma_f32_16x16x32_f16(false, av, false, bv,
                                                        (short)0, acc[t], false, false);
      }
    }

    // bias + GELU + row-sum into pooled (D layout: VGPR r -> m = r + 8*(lane/16), n = lane%16)
    #pragma unroll
    for (int t = 0; t < 8; ++t) {
      const int n = (t << 4) + lmod;
      const float bn = s_b2[n];
      float s = 0.0f;
      #pragma unroll
      for (int r = 0; r < 8; ++r) s += gelu_t(acc[t][r] + bn);
      atomicAdd(&s_pool[n], s);                    // ds_add_f32
    }
  }
  __syncthreads();

  // ---- tail: linear layer3 folded out of the j-sum, then decoder ----
  if (tid < H_DIM) {                               // t3 = pooled @ eW3 + N*eb3
    float s = 1024.0f * eb3[tid];
    for (int k = 0; k < H_DIM; ++k) s += s_pool[k] * eW3[(k << 7) + tid];
    s_tmp[tid] = s;
  }
  __syncthreads();
  if (tid < H_DIM) {                               // d1 = gelu(t3 @ dW1 + db1)
    float s = db1[tid];
    for (int k = 0; k < H_DIM; ++k) s += s_tmp[k] * dW1[(k << 7) + tid];
    s_pool[tid] = gelu_t(s);
  }
  __syncthreads();
  if (tid < H_DIM) {                               // d2 = gelu(d1 @ dW2 + db2)
    float s = db2[tid];
    for (int k = 0; k < H_DIM; ++k) s += s_pool[k] * dW2[(k << 7) + tid];
    s_tmp[tid] = gelu_t(s);
  }
  __syncthreads();
  if (tid < 2) {                                   // out = d2 @ dW3 + db3
    float s = db3[tid];
    for (int k = 0; k < H_DIM; ++k) s += s_tmp[k] * dW3[2*k + tid];
    out[2*i + tid] = s;
  }
}

extern "C" void kernel_launch(void* const* d_in, const int* in_sizes, int n_in,
                              void* d_out, int out_size, void* d_ws, size_t ws_size,
                              hipStream_t stream) {
  const float* xs  = (const float*)d_in[0];
  const float* gs  = (const float*)d_in[1];
  const float* eW1 = (const float*)d_in[2];
  const float* eb1 = (const float*)d_in[3];
  const float* eW2 = (const float*)d_in[4];
  const float* eb2 = (const float*)d_in[5];
  const float* eW3 = (const float*)d_in[6];
  const float* eb3 = (const float*)d_in[7];
  const float* dW1 = (const float*)d_in[8];
  const float* db1 = (const float*)d_in[9];
  const float* dW2 = (const float*)d_in[10];
  const float* db2 = (const float*)d_in[11];
  const float* dW3 = (const float*)d_in[12];
  const float* db3 = (const float*)d_in[13];
  gnn_pair_kernel<<<N_PART, 256, 0, stream>>>(xs, gs, eW1, eb1, eW2, eb2, eW3, eb3,
                                              dW1, db1, dW2, db2, dW3, db3,
                                              (float*)d_out);
}
